// GATv1Layer_81853486727787
// MI455X (gfx1250) — matrix operand order repaired and checked by reading, not compiled, not run on hardware
//
#include <hip/hip_runtime.h>
#include <math.h>

typedef __bf16 bf16_t;
typedef bf16_t v16bf __attribute__((ext_vector_type(16)));
typedef float  v8f   __attribute__((ext_vector_type(8)));

#define C_IN   256
#define C_OUT  256
#define NHEAD  4
#define HDIM   64
#define NEG_SLOPE 0.2f

// ---------------------------------------------------------------------------
// init: zero out/conv/deg/denom, set emax = -inf
// flat index space: [N*256 out][N*8 conv][N deg][N*4 denom][N*4 emax]
// ---------------------------------------------------------------------------
__global__ __launch_bounds__(256) void k_init(float* __restrict__ out,
                                              float* __restrict__ conv,
                                              float* __restrict__ deg,
                                              float* __restrict__ denom,
                                              float* __restrict__ emax, int N) {
  long long i = (long long)blockIdx.x * blockDim.x + threadIdx.x;
  long long t = i;
  if (t < (long long)N * 256) { out[t] = 0.0f; return; }
  t -= (long long)N * 256;
  if (t < (long long)N * 8)   { conv[t] = 0.0f; return; }
  t -= (long long)N * 8;
  if (t < N)                  { deg[t] = 0.0f; return; }
  t -= N;
  if (t < (long long)N * 4)   { denom[t] = 0.0f; return; }
  t -= (long long)N * 4;
  if (t < (long long)N * 4)   { emax[t] = -__builtin_inff(); }
}

// ---------------------------------------------------------------------------
// pack Wv into WMMA-B lane layout, split into bf16 hi/lo.
// B fragment (16x16x32 bf16, wave32): lane = (khalf<<4) | col,
//   element j of the lane's v16bf = B[k0 + khalf*16 + j][col].
// packed unit index: tile t = (k0/32)*16 + (n0/16); elem i = (t*32 + lane)*16 + j
// ---------------------------------------------------------------------------
__global__ __launch_bounds__(256) void k_pack_w(const float* __restrict__ Wv,
                                                bf16_t* __restrict__ Whp,
                                                bf16_t* __restrict__ Wlp) {
  int i = blockIdx.x * blockDim.x + threadIdx.x;   // 0 .. 65535
  if (i >= C_IN * C_OUT) return;
  int j    = i & 15;
  int lane = (i >> 4) & 31;
  int t    = i >> 9;
  int kb   = t >> 4;        // k0/32
  int nt   = t & 15;        // n0/16
  int k = kb * 32 + (lane >> 4) * 16 + j;
  int n = nt * 16 + (lane & 15);
  float f = Wv[k * C_OUT + n];
  bf16_t h = (bf16_t)f;
  Whp[i] = h;
  Wlp[i] = (bf16_t)(f - (float)h);
}

// ---------------------------------------------------------------------------
// fold attention vectors into the weight matrices (exact identity):
//   At[h][c] = sum_d W[c][h*64+d] * att[h][d]   (head-major so scores can float4)
//   sb[h]    = sum_d b[h*64+d]    * att[h][d]
// grid = 8 blocks (side*4 + h), 256 threads (c)
// ---------------------------------------------------------------------------
__global__ __launch_bounds__(256) void k_fold(const float* __restrict__ Wl,
                                              const float* __restrict__ bl,
                                              const float* __restrict__ Wr,
                                              const float* __restrict__ br,
                                              const float* __restrict__ att_l,
                                              const float* __restrict__ att_r,
                                              float* __restrict__ Al,
                                              float* __restrict__ Ar,
                                              float* __restrict__ sb) {
  int side = blockIdx.x >> 2;
  int h    = blockIdx.x & 3;
  int c    = threadIdx.x;
  const float* W    = side ? Wr   : Wl;
  const float* bias = side ? br   : bl;
  const float* att  = side ? att_r : att_l;
  float* A = side ? Ar : Al;
  float acc = 0.0f;
  for (int d = 0; d < HDIM; ++d)
    acc += W[c * C_OUT + h * HDIM + d] * att[h * HDIM + d];
  A[h * C_IN + c] = acc;              // head-major (contiguous per head)
  if (c == 0) {
    float s = 0.0f;
    for (int d = 0; d < HDIM; ++d)
      s += bias[h * HDIM + d] * att[h * HDIM + d];
    sb[side * NHEAD + h] = s;
  }
}

// ---------------------------------------------------------------------------
// xv = x @ Wv + bv via bf16x2-split WMMA: a*b ~= ah*bh + ah*bl + al*bh (fp32 acc)
// block = 256 thr = 8 waves, one 16-row strip; wave w owns columns [w*32, w*32+32)
// as two 16x16 accumulators. 8 k-steps of 32 -> 48 v_wmma per wave.
// ---------------------------------------------------------------------------
__global__ __launch_bounds__(256) void k_gemm_xv(const float* __restrict__ x,
                                                 const bf16_t* __restrict__ Whp,
                                                 const bf16_t* __restrict__ Wlp,
                                                 const float* __restrict__ bv,
                                                 float* __restrict__ xv, int N) {
  const int m0    = blockIdx.x * 16;
  const int wave  = threadIdx.x >> 5;
  const int lane  = threadIdx.x & 31;
  const int row   = lane & 15;
  const int khalf = lane >> 4;
  const int n0    = wave * 32;

  int mload = m0 + row; if (mload >= N) mload = N - 1;   // tail clamp (reads only)
  const float* xrow = x + (long long)mload * C_IN;
  const v16bf* Bh = (const v16bf*)Whp;
  const v16bf* Bl = (const v16bf*)Wlp;

  v8f acc0 = {0.f,0.f,0.f,0.f,0.f,0.f,0.f,0.f};
  v8f acc1 = {0.f,0.f,0.f,0.f,0.f,0.f,0.f,0.f};

  for (int k0 = 0; k0 < C_IN; k0 += 32) {
    // A fragment (16x32 bf16): lanes 0-15 hold K {k0..k0+15},
    // lanes 16-31 hold K {k0+16..k0+31}; hi/lo split in-register.
    v16bf ah, al;
#pragma unroll
    for (int j = 0; j < 8; ++j) {
      float f0 = xrow[k0 + khalf * 16 + j];
      float f1 = xrow[k0 + khalf * 16 + 8 + j];
      bf16_t h0 = (bf16_t)f0, h1 = (bf16_t)f1;
      ah[j]     = h0;  al[j]     = (bf16_t)(f0 - (float)h0);
      ah[j + 8] = h1;  al[j + 8] = (bf16_t)(f1 - (float)h1);
    }
    __builtin_prefetch(xrow + k0 + 32, 0, 0);   // global_prefetch next K slab

    const int t0 = ((k0 >> 5) * 16 + (n0 >> 4)) * 32 + lane;
    const int t1 = t0 + 32;                     // next 16-column tile
    v16bf bh0 = Bh[t0], bl0 = Bl[t0];
    v16bf bh1 = Bh[t1], bl1 = Bl[t1];

    acc0 = __builtin_amdgcn_wmma_f32_16x16x32_bf16(false, ah, false, bh0, (short)0, acc0, false, false);
    acc1 = __builtin_amdgcn_wmma_f32_16x16x32_bf16(false, ah, false, bh1, (short)0, acc1, false, false);
    acc0 = __builtin_amdgcn_wmma_f32_16x16x32_bf16(false, ah, false, bl0, (short)0, acc0, false, false);
    acc1 = __builtin_amdgcn_wmma_f32_16x16x32_bf16(false, ah, false, bl1, (short)0, acc1, false, false);
    acc0 = __builtin_amdgcn_wmma_f32_16x16x32_bf16(false, al, false, bh0, (short)0, acc0, false, false);
    acc1 = __builtin_amdgcn_wmma_f32_16x16x32_bf16(false, al, false, bh1, (short)0, acc1, false, false);
  }

  // C/D layout: VGPR r -> M = r + 8*khalf, N = lane&15
  const int mbase = m0 + 8 * khalf;
  const float b0 = bv[n0 + row];
  const float b1 = bv[n0 + 16 + row];
  if (m0 + 16 <= N) {
    // uniform fast path: whole tile in range, unguarded stores
#pragma unroll
    for (int r = 0; r < 8; ++r) {
      long long o = (long long)(mbase + r) * C_OUT;
      xv[o + n0 + row]      = acc0[r] + b0;
      xv[o + n0 + 16 + row] = acc1[r] + b1;
    }
  } else {
#pragma unroll
    for (int r = 0; r < 8; ++r) {
      int mm = mbase + r;
      if (mm < N) {
        long long o = (long long)mm * C_OUT;
        xv[o + n0 + row]      = acc0[r] + b0;
        xv[o + n0 + 16 + row] = acc1[r] + b1;
      }
    }
  }
}

// ---------------------------------------------------------------------------
// hlr[n][s] = x[n] . At[h] + sb   (s<4: left, s>=4: right), float4 GEMV
// ---------------------------------------------------------------------------
__global__ __launch_bounds__(256) void k_scores(const float* __restrict__ x,
                                                const float* __restrict__ Al,
                                                const float* __restrict__ Ar,
                                                const float* __restrict__ sb,
                                                float* __restrict__ hlr, int N) {
  long long i = (long long)blockIdx.x * blockDim.x + threadIdx.x;
  if (i >= (long long)N * 8) return;
  int n = (int)(i >> 3);
  int s = (int)(i & 7);
  int h = s & 3;
  const float* A  = ((s < 4) ? Al : Ar) + h * C_IN;   // contiguous 256 floats
  const float* xr = x + (long long)n * C_IN;
  float acc = 0.0f;
  for (int c = 0; c < C_IN; c += 4) {
    float4 xa = *(const float4*)(xr + c);
    float4 aa = *(const float4*)(A + c);
    acc += xa.x * aa.x + xa.y * aa.y + xa.z * aa.z + xa.w * aa.w;
  }
  hlr[i] = acc + sb[s];
}

// ---------------------------------------------------------------------------
// conv = segment_sum(hlr[src], dst); deg = segment_sum(1, dst)
// ---------------------------------------------------------------------------
__global__ __launch_bounds__(256) void k_edge_conv(const int* __restrict__ src,
                                                   const int* __restrict__ dst,
                                                   const float* __restrict__ hlr,
                                                   float* __restrict__ conv,
                                                   float* __restrict__ deg, int E) {
  long long i = (long long)blockIdx.x * blockDim.x + threadIdx.x;
  if (i >= (long long)E * 8) return;
  int e = (int)(i >> 3);
  int s = (int)(i & 7);
  int sn = src[e], dn = dst[e];
  atomicAdd(&conv[(long long)dn * 8 + s], hlr[(long long)sn * 8 + s]);
  if (s == 0) atomicAdd(&deg[dn], 1.0f);
}

// agg = (hlr + conv) / (1 + deg), in place over hlr
__global__ __launch_bounds__(256) void k_agg(float* __restrict__ hlr,
                                             const float* __restrict__ conv,
                                             const float* __restrict__ deg, int N) {
  long long i = (long long)blockIdx.x * blockDim.x + threadIdx.x;
  if (i >= (long long)N * 8) return;
  int n = (int)(i >> 3);
  hlr[i] = (hlr[i] + conv[i]) / (1.0f + deg[n]);
}

__device__ inline void atomicMaxFloat(float* addr, float v) {
  // sign-split trick: correct for mixed-sign values, init = -inf
  if (v >= 0.0f) atomicMax((int*)addr, __float_as_int(v));
  else           atomicMin((unsigned int*)addr, __float_as_uint(v));
}

__device__ inline float edge_score(const float* hlr, int sn, int dn, int h) {
  float v = hlr[(long long)sn * 8 + h] + hlr[(long long)dn * 8 + 4 + h];
  return v > 0.0f ? v : NEG_SLOPE * v;   // leaky_relu
}

// emax = segment_max(e, dst)
__global__ __launch_bounds__(256) void k_edge_max(const int* __restrict__ src,
                                                  const int* __restrict__ dst,
                                                  const float* __restrict__ hlr,
                                                  float* __restrict__ emax, int E) {
  long long i = (long long)blockIdx.x * blockDim.x + threadIdx.x;
  if (i >= (long long)E * 4) return;
  int e = (int)(i >> 2);
  int h = (int)(i & 3);
  int dn = dst[e];
  atomicMaxFloat(&emax[(long long)dn * 4 + h], edge_score(hlr, src[e], dn, h));
}

// emax = where(isfinite(emax), emax, 0)
__global__ __launch_bounds__(256) void k_fix_emax(float* __restrict__ emax, int N) {
  long long i = (long long)blockIdx.x * blockDim.x + threadIdx.x;
  if (i >= (long long)N * 4) return;
  float v = emax[i];
  if (!(v > -1e37f)) emax[i] = 0.0f;
}

// ex = exp(e - emax[dst]); denom = segment_sum(ex, dst)
__global__ __launch_bounds__(256) void k_edge_exp(const int* __restrict__ src,
                                                  const int* __restrict__ dst,
                                                  const float* __restrict__ hlr,
                                                  const float* __restrict__ emax,
                                                  float* __restrict__ ex,
                                                  float* __restrict__ denom, int E) {
  long long i = (long long)blockIdx.x * blockDim.x + threadIdx.x;
  if (i >= (long long)E * 4) return;
  int e = (int)(i >> 2);
  int h = (int)(i & 3);
  int dn = dst[e];
  float v  = edge_score(hlr, src[e], dn, h);
  float xv = expf(v - emax[(long long)dn * 4 + h]);
  ex[i] = xv;
  atomicAdd(&denom[(long long)dn * 4 + h], xv);
}

// out += xv[src] * (ex / max(denom[dst], 1e-16))   -- bandwidth-dominant phase.
// One thread per (edge, 4-channel group): b128 gather + 4 f32 atomics.
__global__ __launch_bounds__(256) void k_edge_out(const int* __restrict__ src,
                                                  const int* __restrict__ dst,
                                                  const float* __restrict__ xv,
                                                  const float* __restrict__ ex,
                                                  const float* __restrict__ denom,
                                                  float* __restrict__ out, int E) {
  long long i = (long long)blockIdx.x * blockDim.x + threadIdx.x;
  if (i >= (long long)E * 64) return;
  int e = (int)(i >> 6);
  int q = (int)(i & 63);       // 4-channel group
  int c = q << 2;
  int h = q >> 4;              // head = c/64
  int sn = src[e], dn = dst[e];
  float a = ex[(long long)e * 4 + h] / fmaxf(denom[(long long)dn * 4 + h], 1e-16f);
  float4 v = *(const float4*)(xv + (long long)sn * 256 + c);
  float* o = out + (long long)dn * 256 + c;
  atomicAdd(o + 0, v.x * a);
  atomicAdd(o + 1, v.y * a);
  atomicAdd(o + 2, v.z * a);
  atomicAdd(o + 3, v.w * a);
}

// ---------------------------------------------------------------------------
extern "C" void kernel_launch(void* const* d_in, const int* in_sizes, int n_in,
                              void* d_out, int out_size, void* d_ws, size_t ws_size,
                              hipStream_t stream) {
  const float* x     = (const float*)d_in[0];
  const int*   src   = (const int*)d_in[1];
  const int*   dst   = (const int*)d_in[2];
  const float* Wl    = (const float*)d_in[3];
  const float* bl    = (const float*)d_in[4];
  const float* Wr    = (const float*)d_in[5];
  const float* br    = (const float*)d_in[6];
  const float* Wv    = (const float*)d_in[7];
  const float* bv    = (const float*)d_in[8];
  const float* att_l = (const float*)d_in[9];
  const float* att_r = (const float*)d_in[10];
  float* out = (float*)d_out;

  const int N = in_sizes[0] / C_IN;   // 50000
  const int E = in_sizes[1];          // 800000

  // workspace layout
  char* ws = (char*)d_ws;
  size_t o_xv    = 0;                                   // N*256 f32
  size_t o_hlr   = o_xv    + (size_t)N * 256 * 4;       // N*8 f32
  size_t o_conv  = o_hlr   + (size_t)N * 8 * 4;         // N*8 f32
  size_t o_deg   = o_conv  + (size_t)N * 8 * 4;         // N f32
  size_t o_emax  = o_deg   + (size_t)N * 4;             // N*4 f32
  size_t o_denom = o_emax  + (size_t)N * 4 * 4;         // N*4 f32
  size_t o_ex    = o_denom + (size_t)N * 4 * 4;         // E*4 f32
  size_t o_Al    = o_ex    + (size_t)E * 4 * 4;         // 4*256 f32 (head-major)
  size_t o_Ar    = o_Al    + 256 * 4 * 4;               // 4*256 f32
  size_t o_sb    = o_Ar    + 256 * 4 * 4;               // 8 f32
  size_t o_Wh    = o_sb    + 256;                       // 64K bf16
  size_t o_Wlo   = o_Wh    + (size_t)C_IN * C_OUT * 2;  // 64K bf16

  float*  xv_   = (float*)(ws + o_xv);
  float*  hlr   = (float*)(ws + o_hlr);
  float*  conv  = (float*)(ws + o_conv);
  float*  deg   = (float*)(ws + o_deg);
  float*  emax  = (float*)(ws + o_emax);
  float*  denom = (float*)(ws + o_denom);
  float*  ex    = (float*)(ws + o_ex);
  float*  Al    = (float*)(ws + o_Al);
  float*  Ar    = (float*)(ws + o_Ar);
  float*  sb    = (float*)(ws + o_sb);
  bf16_t* Whp   = (bf16_t*)(ws + o_Wh);
  bf16_t* Wlp   = (bf16_t*)(ws + o_Wlo);

  const int TPB = 256;
  auto blk = [](long long n, int tpb) { return (unsigned)((n + tpb - 1) / tpb); };

  long long init_total = (long long)N * 273;  // 256+8+1+4+4
  k_init<<<blk(init_total, TPB), TPB, 0, stream>>>(out, conv, deg, denom, emax, N);
  k_pack_w<<<blk(C_IN * C_OUT, TPB), TPB, 0, stream>>>(Wv, Whp, Wlp);
  k_fold<<<8, TPB, 0, stream>>>(Wl, bl, Wr, br, att_l, att_r, Al, Ar, sb);
  k_gemm_xv<<<blk(N, 16), TPB, 0, stream>>>(x, Whp, Wlp, bv, xv_, N);
  k_scores<<<blk((long long)N * 8, TPB), TPB, 0, stream>>>(x, Al, Ar, sb, hlr, N);
  k_edge_conv<<<blk((long long)E * 8, TPB), TPB, 0, stream>>>(src, dst, hlr, conv, deg, E);
  k_agg<<<blk((long long)N * 8, TPB), TPB, 0, stream>>>(hlr, conv, deg, N);
  k_edge_max<<<blk((long long)E * 4, TPB), TPB, 0, stream>>>(src, dst, hlr, emax, E);
  k_fix_emax<<<blk((long long)N * 4, TPB), TPB, 0, stream>>>(emax, N);
  k_edge_exp<<<blk((long long)E * 4, TPB), TPB, 0, stream>>>(src, dst, hlr, emax, ex, denom, E);
  k_edge_out<<<blk((long long)E * 64, TPB), TPB, 0, stream>>>(src, dst, xv_, ex, denom, out, E);
  (void)n_in; (void)out_size; (void)ws_size;
}
